// BahdanauAttention_15015205667169
// MI455X (gfx1250) — compile-verified, matrix operand order checked by reading, not run on hardware
//
#include <hip/hip_runtime.h>
#include <hip/hip_bf16.h>

typedef __attribute__((ext_vector_type(16))) __bf16 v16bf;
typedef __attribute__((ext_vector_type(8)))  __bf16 v8bf;
typedef __attribute__((ext_vector_type(4)))  __bf16 v4bf;
typedef __attribute__((ext_vector_type(8)))  float  v8f;

#define BATCH    32
#define SEQ      2048
#define ENC_DIM  1024
#define ATTN_DIM 1024

// Hardware tanh on gfx1250 (V_TANH_F32, TRANS32 op) with safe fallback.
__device__ __forceinline__ float fast_tanh(float x) {
#if __has_builtin(__builtin_amdgcn_tanhf)
    return __builtin_amdgcn_tanhf(x);
#else
    return tanhf(x);
#endif
}

// ---------------------------------------------------------------------------
// Kernel 0: zero the context region of d_out (context kernel accumulates with
// atomics).
// ---------------------------------------------------------------------------
__global__ void zero_f32(float* __restrict__ p, int n) {
    int i = blockIdx.x * blockDim.x + threadIdx.x;
    if (i < n) p[i] = 0.0f;
}

// ---------------------------------------------------------------------------
// Kernel 1: W_enc (E x A, f32, row-major) -> Wt (A x E, bf16, row-major).
// N-major bf16 weights make every WMMA B-fragment a contiguous 16B load.
// ---------------------------------------------------------------------------
__global__ void wenc_transpose_bf16(const float* __restrict__ W,
                                    __bf16* __restrict__ Wt) {
    __shared__ float tile[32][33];
    const int a0 = blockIdx.x * 32;
    const int e0 = blockIdx.y * 32;
    const int tx = threadIdx.x, ty = threadIdx.y;
    tile[ty][tx] = W[(size_t)(e0 + ty) * ATTN_DIM + a0 + tx];
    __syncthreads();
    Wt[(size_t)(a0 + ty) * ENC_DIM + e0 + tx] = (__bf16)tile[tx][ty];
}

// ---------------------------------------------------------------------------
// Kernel 2: dec_proj[b][a] = decoder_hidden[b,:] . W_dec[:,a] + b_dec[a] + b_enc[a]
// (b_enc folded in so the fused kernel adds a single bias vector).
// ---------------------------------------------------------------------------
__global__ void dec_proj_kernel(const float* __restrict__ dh,
                                const float* __restrict__ Wd,
                                const float* __restrict__ bd,
                                const float* __restrict__ be,
                                float* __restrict__ dp) {
    const int a = blockIdx.y * 256 + threadIdx.x;
    const int b = blockIdx.x;
    const float* dhb = dh + (size_t)b * ENC_DIM;
    float acc = 0.0f;
    #pragma unroll 4
    for (int k = 0; k < ENC_DIM; ++k)
        acc += dhb[k] * Wd[(size_t)k * ATTN_DIM + a];
    dp[(size_t)b * ATTN_DIM + a] = acc + bd[a] + be[a];
}

// ---------------------------------------------------------------------------
// Kernel 3 (the heavy one): fused
//   scores[b,s] = sum_a tanh( (enc[b,s,:] @ W_enc)[a] + dp[b,a] ) * v[a]
// Block = 512 threads (16 waves), M-tile = 64 rows of flattened (b,s).
// Wave w owns N-columns [64w, 64w+64): 4 N-tiles x 4 M-tiles of
// v_wmma_f32_16x16x32_bf16, f32 accumulate.  Encoder tiles are converted
// f32->bf16 through a double-buffered LDS stage (one barrier per k-step);
// B comes straight from the transposed bf16 weights (L2-resident, 2 MB).
// Accumulators never leave VGPRs: hardware tanh + dot-with-v + shfl_xor
// reduction collapse each 16x16 tile row to a scalar.
// ---------------------------------------------------------------------------
__launch_bounds__(512)
__global__ void fused_scores_kernel(const float* __restrict__ enc,
                                    const __bf16* __restrict__ Wt,
                                    const float* __restrict__ dp,
                                    const float* __restrict__ vw,
                                    const float* __restrict__ vb,
                                    float* __restrict__ scores) {
    const int tid  = threadIdx.x;
    const int wave = tid >> 5;          // 0..15
    const int lane = tid & 31;
    const int l16  = lane & 15;
    const int hi   = lane >> 4;         // 0: lanes 0-15, 1: lanes 16-31
    const int bs0  = blockIdx.x * 64;   // first flattened (b,s) row
    const int b    = bs0 >> 11;         // bs0 / 2048 (tiles never straddle b)

    // Double-buffered A stage: 2 x (64 rows x 32 K) bf16, row stride padded
    // to 40 halfs (80 B) to spread LDS banks.
    __shared__ __bf16 As[2][64 * 40];
    __shared__ float  ssc[64];

    if (tid < 64) ssc[tid] = 0.0f;

    union Frag { v16bf v; v8bf h[2]; };

    const v8f vzero = {0.f, 0.f, 0.f, 0.f, 0.f, 0.f, 0.f, 0.f};
    v8f acc[4][4];
    #pragma unroll
    for (int m = 0; m < 4; ++m)
        #pragma unroll
        for (int nt = 0; nt < 4; ++nt)
            acc[m][nt] = vzero;

    const int nbase = wave * 64;
    const float* encBase = enc + (size_t)bs0 * ENC_DIM;
    const int kb = hi << 3;             // per-lane K sub-offset: 0 or 8

    // Per-thread staging coordinates (512 threads x 4 f32 = 64x32 tile).
    const int sidx = tid << 2;
    const int srow = sidx >> 5;
    const int skk  = sidx & 31;
    const float* sld = encBase + (size_t)srow * ENC_DIM + skk;

    auto stage = [&](int buf, int k0) {
        const float4 f = *(const float4*)(sld + k0);
        v4bf h;
        h.x = (__bf16)f.x; h.y = (__bf16)f.y;
        h.z = (__bf16)f.z; h.w = (__bf16)f.w;
        *(v4bf*)&As[buf][srow * 40 + skk] = h;
    };

    stage(0, 0);

    for (int k0 = 0; k0 < ENC_DIM; k0 += 32) {
        const int cur = (k0 >> 5) & 1;
        __syncthreads();                      // staged tile `cur` visible
        if (k0 + 32 < ENC_DIM)
            stage(cur ^ 1, k0 + 32);          // overlaps with WMMAs below

        // ---- A fragments (4 M sub-tiles) from LDS ----
        Frag afrag[4];
        #pragma unroll
        for (int m = 0; m < 4; ++m) {
            const __bf16* ap = &As[cur][(m * 16 + l16) * 40];
            afrag[m].h[0] = *(const v8bf*)(ap + kb);        // K = kb .. kb+7
            afrag[m].h[1] = *(const v8bf*)(ap + kb + 16);   // K = kb+16 .. kb+23
        }

        // ---- B fragments from bf16 N-major weights + 16 WMMAs ----
        #pragma unroll
        for (int nt = 0; nt < 4; ++nt) {
            const __bf16* bp = Wt + (size_t)(nbase + nt * 16 + l16) * ENC_DIM + k0 + kb;
            Frag bfrag;
            bfrag.h[0] = *(const v8bf*)(bp);
            bfrag.h[1] = *(const v8bf*)(bp + 16);
            #pragma unroll
            for (int m = 0; m < 4; ++m)
                acc[m][nt] = __builtin_amdgcn_wmma_f32_16x16x32_bf16(
                    false, afrag[m].v, false, bfrag.v,
                    (short)0, acc[m][nt], false, false);
        }
    }

    // ---- epilogue: tanh(acc + dp) . v, reduced to one scalar per row ----
    const float* dpb = dp + (size_t)b * ATTN_DIM;
    float dn[4], wn[4];
    #pragma unroll
    for (int nt = 0; nt < 4; ++nt) {
        const int n = nbase + nt * 16 + l16;
        dn[nt] = dpb[n];
        wn[nt] = vw[n];
    }

    #pragma unroll
    for (int m = 0; m < 4; ++m) {
        #pragma unroll
        for (int j = 0; j < 8; ++j) {
            float part = 0.0f;
            #pragma unroll
            for (int nt = 0; nt < 4; ++nt)
                part += fast_tanh(acc[m][nt][j] + dn[nt]) * wn[nt];
            // reduce across the 16 lanes of this half-wave (C layout: lanes
            // 0-15 hold row j, lanes 16-31 hold row j+8)
            part += __shfl_xor(part, 1);
            part += __shfl_xor(part, 2);
            part += __shfl_xor(part, 4);
            part += __shfl_xor(part, 8);
            if (l16 == 0) {
                const int row = m * 16 + j + (hi << 3);
                atomicAdd(&ssc[row], part);     // ds_add_f32 across 16 waves
            }
        }
    }
    __syncthreads();
    if (tid < 64) scores[bs0 + tid] = ssc[tid] + vb[0];
}

// ---------------------------------------------------------------------------
// Kernel 4: softmax over S=2048 per batch row.
// ---------------------------------------------------------------------------
__global__ void softmax_kernel(const float* __restrict__ scores,
                               float* __restrict__ weights) {
    const int b = blockIdx.x;
    const int tid = threadIdx.x;            // 256
    __shared__ float red[256];
    const float* sr = scores + (size_t)b * SEQ;

    float v[8];
    float m = -INFINITY;
    #pragma unroll
    for (int i = 0; i < 8; ++i) {
        v[i] = sr[tid + i * 256];
        m = fmaxf(m, v[i]);
    }
    red[tid] = m; __syncthreads();
    for (int off = 128; off > 0; off >>= 1) {
        if (tid < off) red[tid] = fmaxf(red[tid], red[tid + off]);
        __syncthreads();
    }
    m = red[0];
    __syncthreads();

    float s = 0.0f;
    #pragma unroll
    for (int i = 0; i < 8; ++i) {
        v[i] = __expf(v[i] - m);
        s += v[i];
    }
    red[tid] = s; __syncthreads();
    for (int off = 128; off > 0; off >>= 1) {
        if (tid < off) red[tid] += red[tid + off];
        __syncthreads();
    }
    const float inv = 1.0f / red[0];
    #pragma unroll
    for (int i = 0; i < 8; ++i)
        weights[(size_t)b * SEQ + tid + i * 256] = v[i] * inv;
}

// ---------------------------------------------------------------------------
// Kernel 5: context[b,e] = sum_s w[b,s] * enc[b,s,e].  Memory-bound streaming
// of the 256 MB encoder tensor; grid.y splits S for parallelism, partial sums
// merged with f32 global atomics.
// ---------------------------------------------------------------------------
__global__ void context_kernel(const float* __restrict__ enc,
                               const float* __restrict__ weights,
                               float* __restrict__ ctx) {
    const int b     = blockIdx.x;
    const int chunk = blockIdx.y;           // 8 chunks of 256 s each
    const int tid   = threadIdx.x;          // 256 threads -> 4 e each = 1024
    const float* w = weights + (size_t)b * SEQ + chunk * 256;
    const float* e = enc + (size_t)b * SEQ * ENC_DIM + (size_t)chunk * 256 * ENC_DIM;

    float4 acc = {0.f, 0.f, 0.f, 0.f};
    for (int s = 0; s < 256; ++s) {
        const float ws = w[s];
        const float4 x = *(const float4*)(e + (size_t)s * ENC_DIM + tid * 4);
        acc.x += ws * x.x; acc.y += ws * x.y;
        acc.z += ws * x.z; acc.w += ws * x.w;
    }
    float* c = ctx + (size_t)b * ENC_DIM + tid * 4;
    atomicAdd(c + 0, acc.x);
    atomicAdd(c + 1, acc.y);
    atomicAdd(c + 2, acc.z);
    atomicAdd(c + 3, acc.w);
}

// ---------------------------------------------------------------------------
extern "C" void kernel_launch(void* const* d_in, const int* in_sizes, int n_in,
                              void* d_out, int out_size, void* d_ws, size_t ws_size,
                              hipStream_t stream) {
    const float* enc = (const float*)d_in[0];   // (32, 2048, 1024)
    const float* dh  = (const float*)d_in[1];   // (32, 1024)
    const float* We  = (const float*)d_in[2];   // (1024, 1024)
    const float* be  = (const float*)d_in[3];   // (1024,)
    const float* Wd  = (const float*)d_in[4];   // (1024, 1024)
    const float* bd  = (const float*)d_in[5];   // (1024,)
    const float* vw  = (const float*)d_in[6];   // (1024,)
    const float* vb  = (const float*)d_in[7];   // (1,)

    float* ctx = (float*)d_out;                         // 32*1024
    float* wts = (float*)d_out + BATCH * ENC_DIM;       // 32*2048

    char* ws = (char*)d_ws;
    __bf16* Wt     = (__bf16*)ws;                                      // 2 MB
    float*  dp     = (float*)(ws + 2u * 1024u * 1024u);                // 128 KB
    float*  scores = (float*)(ws + 2u * 1024u * 1024u + 128u * 1024u); // 256 KB

    zero_f32<<<(BATCH * ENC_DIM + 255) / 256, 256, 0, stream>>>(ctx, BATCH * ENC_DIM);

    wenc_transpose_bf16<<<dim3(ATTN_DIM / 32, ENC_DIM / 32), dim3(32, 32), 0, stream>>>(We, Wt);

    dec_proj_kernel<<<dim3(BATCH, ATTN_DIM / 256), 256, 0, stream>>>(dh, Wd, bd, be, dp);

    fused_scores_kernel<<<(BATCH * SEQ) / 64, 512, 0, stream>>>(enc, Wt, dp, vw, vb, scores);

    softmax_kernel<<<BATCH, 256, 0, stream>>>(scores, wts);

    context_kernel<<<dim3(BATCH, 8), 256, 0, stream>>>(enc, wts, ctx);
}
